// LoadForecastingModel_85882166050869
// MI455X (gfx1250) — matrix operand order repaired and checked by reading, not compiled, not run on hardware
//
#include <hip/hip_runtime.h>
#include <hip/hip_bf16.h>

typedef __attribute__((ext_vector_type(16))) __bf16 v16bf;
typedef __attribute__((ext_vector_type(8)))  __bf16 v8bf;
typedef __attribute__((ext_vector_type(8)))  float  v8f;

#define T_STEPS 168
#define I_SZ 32
#define H_SZ 128

// Branch-free activations on the native TRANS pipe (v_exp_f32 / v_rcp_f32).
// Saturate correctly at +/-inf; precision ~1 ulp of native ops, feeding bf16
// state anyway so full-precision OCML expf (40+ VALU ops + divergent
// branches) is wasted latency on the serial recurrence path.
__device__ __forceinline__ float fast_sigmoid(float v) {
    float e = __builtin_amdgcn_exp2f(v * -1.4426950408889634f);
    return __builtin_amdgcn_rcpf(1.0f + e);
}
__device__ __forceinline__ float fast_tanh(float v) {
    float e = __builtin_amdgcn_exp2f(v * -2.8853900817779268f);
    return __builtin_amdgcn_rcpf(1.0f + e) * 2.0f - 1.0f;
}

__global__ __launch_bounds__(256, 1)
void lstm_fused_kernel(const float* __restrict__ x,
                       const float* __restrict__ w_ih,
                       const float* __restrict__ w_hh,
                       const float* __restrict__ b_ih,
                       const float* __restrict__ b_hh,
                       const float* __restrict__ w1,
                       const float* __restrict__ b1,
                       const float* __restrict__ w2,
                       const float* __restrict__ b2,
                       const float* __restrict__ w3,
                       const float* __restrict__ b3,
                       float* __restrict__ out)
{
    // h tile exchanged across waves every timestep (bf16, A-layout source)
    __shared__ __bf16 h_lds[16][H_SZ];   // 4 KB
    __shared__ float  hf32[16][H_SZ];    // 8 KB, final hidden state for MLP
    __shared__ float  h1s[16][64];       // 4 KB
    __shared__ float  h2s[16][32];       // 2 KB

    const int tid  = threadIdx.x;
    const int wave = tid >> 5;           // 0..7  (wave32)
    const int lane = tid & 31;
    const int m    = lane & 15;          // row / col index within 16
    const int hi   = lane >> 4;          // lane half
    const int b0   = blockIdx.x * 16;    // batch tile base

    // Wave w owns gate columns [16w, 16w+16) of each gate i/f/g/o.
    const int ncol = wave * 16 + m;      // column within one gate (0..127)

    // ---------------- load weight B-fragments (register resident) -------------
    // B-frag element e: k = hi*16 + e (within 32-K block), col n = m of tile.
    // B[k][n] = W^T[k][nglob] = W[nglob][k]
    v16bf whh[4][4];   // [kblock][gate]
    v16bf wih[4];      // [gate]  (K = 32 = full I)
    float bsum[4];
    #pragma unroll
    for (int g = 0; g < 4; ++g) {
        const int nglob = g * H_SZ + ncol;            // row of w_hh / w_ih
        const float* wr = w_hh + (size_t)nglob * H_SZ;
        #pragma unroll
        for (int kb = 0; kb < 4; ++kb) {
            const float* p = wr + kb * 32 + hi * 16;
            v16bf f;
            #pragma unroll
            for (int e = 0; e < 16; ++e) f[e] = (__bf16)p[e];
            whh[kb][g] = f;
        }
        const float* q = w_ih + (size_t)nglob * I_SZ + hi * 16;
        v16bf fx;
        #pragma unroll
        for (int e = 0; e < 16; ++e) fx[e] = (__bf16)q[e];
        wih[g] = fx;
        bsum[g] = b_ih[g * H_SZ + ncol] + b_hh[g * H_SZ + ncol];
    }

    // ---------------- init h = 0, c = 0 --------------------------------------
    for (int idx = tid; idx < 16 * H_SZ; idx += 256)
        ((__bf16*)h_lds)[idx] = (__bf16)0.0f;
    float c[8];
    #pragma unroll
    for (int r = 0; r < 8; ++r) c[r] = 0.0f;
    __syncthreads();

    // A-fragment K offset for this lane: k = basek + e
    const int basek = hi * 16;
    const float* xrow = x + (size_t)(b0 + m) * T_STEPS * I_SZ;

    // ---------------- time loop ------------------------------------------------
    for (int t = 0; t < T_STEPS; ++t) {
        const float* xp = xrow + (size_t)t * I_SZ;
        // Stream prefetch of x two steps ahead (speculative, drops past end).
        __builtin_prefetch(xp + 2 * I_SZ, 0, 1);

        // Issue all h A-fragment LDS loads first; x-projection WMMAs below
        // execute while these return (no per-kb dscnt stall).
        v16bf ha[4];
        #pragma unroll
        for (int kb = 0; kb < 4; ++kb) {
            v8bf lo = *(const v8bf*)&h_lds[m][kb * 32 + basek];
            v8bf hh = *(const v8bf*)&h_lds[m][kb * 32 + basek + 8];
            v16bf f;
            #pragma unroll
            for (int e = 0; e < 8; ++e) { f[e] = lo[e]; f[8 + e] = hh[e]; }
            ha[kb] = f;
        }

        // x A-fragment: row m, K = 0..31, one contiguous 16-float run per lane
        float4 a0 = *(const float4*)(xp + basek);
        float4 a1 = *(const float4*)(xp + basek + 4);
        float4 a2 = *(const float4*)(xp + basek + 8);
        float4 a3 = *(const float4*)(xp + basek + 12);
        float xv[16] = {a0.x, a0.y, a0.z, a0.w, a1.x, a1.y, a1.z, a1.w,
                        a2.x, a2.y, a2.z, a2.w, a3.x, a3.y, a3.z, a3.w};
        v16bf xa;
        #pragma unroll
        for (int e = 0; e < 16; ++e) xa[e] = (__bf16)xv[e];

        // gates = bias + x @ W_ih^T + h @ W_hh^T   (acc in f32 C-layout)
        v8f acc[4];
        #pragma unroll
        for (int g = 0; g < 4; ++g) {
            v8f a;
            #pragma unroll
            for (int e = 0; e < 8; ++e) a[e] = bsum[g];
            acc[g] = __builtin_amdgcn_wmma_f32_16x16x32_bf16(
                false, xa, false, wih[g], (short)0, a, false, false);
        }
        #pragma unroll
        for (int kb = 0; kb < 4; ++kb) {
            #pragma unroll
            for (int g = 0; g < 4; ++g) {
                acc[g] = __builtin_amdgcn_wmma_f32_16x16x32_bf16(
                    false, ha[kb], false, whh[kb][g], (short)0, acc[g], false, false);
            }
        }

        // elementwise LSTM cell update (C-layout: reg r -> row r+8*hi, col ncol)
        float hnew[8];
        #pragma unroll
        for (int r = 0; r < 8; ++r) {
            float iv = fast_sigmoid(acc[0][r]);
            float fv = fast_sigmoid(acc[1][r]);
            float gv = fast_tanh(acc[2][r]);
            float ov = fast_sigmoid(acc[3][r]);
            float cv = fv * c[r] + iv * gv;
            c[r] = cv;
            hnew[r] = ov * fast_tanh(cv);
        }

        __syncthreads();   // all waves finished reading previous h
        #pragma unroll
        for (int r = 0; r < 8; ++r) {
            const int row = r + hi * 8;
            h_lds[row][ncol] = (__bf16)hnew[r];
            if (t == T_STEPS - 1) hf32[row][ncol] = hnew[r];
        }
        __syncthreads();   // new h visible before next step's reads
    }

    // ---------------- MLP head: 128 -> 64 -> 32 -> 1 ---------------------------
    for (int idx = tid; idx < 16 * 64; idx += 256) {
        const int row = idx >> 6, u = idx & 63;
        float s = b1[u];
        const float* wr = w1 + (size_t)u * H_SZ;
        #pragma unroll 8
        for (int k = 0; k < H_SZ; ++k) s += hf32[row][k] * wr[k];
        h1s[row][u] = fmaxf(s, 0.0f);
    }
    __syncthreads();
    for (int idx = tid; idx < 16 * 32; idx += 256) {
        const int row = idx >> 5, u = idx & 31;
        float s = b2[u];
        const float* wr = w2 + (size_t)u * 64;
        #pragma unroll 8
        for (int k = 0; k < 64; ++k) s += h1s[row][k] * wr[k];
        h2s[row][u] = fmaxf(s, 0.0f);
    }
    __syncthreads();
    if (tid < 16) {
        float s = b3[0];
        #pragma unroll
        for (int k = 0; k < 32; ++k) s += h2s[tid][k] * w3[k];
        out[b0 + tid] = s;
    }
}

extern "C" void kernel_launch(void* const* d_in, const int* in_sizes, int n_in,
                              void* d_out, int out_size, void* d_ws, size_t ws_size,
                              hipStream_t stream) {
    const float* x    = (const float*)d_in[0];
    const float* w_ih = (const float*)d_in[1];
    const float* w_hh = (const float*)d_in[2];
    const float* b_ih = (const float*)d_in[3];
    const float* b_hh = (const float*)d_in[4];
    const float* w1   = (const float*)d_in[5];
    const float* b1   = (const float*)d_in[6];
    const float* w2   = (const float*)d_in[7];
    const float* b2   = (const float*)d_in[8];
    const float* w3   = (const float*)d_in[9];
    const float* b3   = (const float*)d_in[10];
    float* out = (float*)d_out;

    const int batch = in_sizes[0] / (T_STEPS * I_SZ);   // 4096
    const int grid  = batch / 16;                       // 256 workgroups
    lstm_fused_kernel<<<grid, 256, 0, stream>>>(x, w_ih, w_hh, b_ih, b_hh,
                                                w1, b1, w2, b2, w3, b3, out);
}
